// Model_47579647705453
// MI455X (gfx1250) — matrix operand order repaired and checked by reading, not compiled, not run on hardware
//
#include <hip/hip_runtime.h>

#define GAMMA_F 0.4f
#define VOCAB   32000
#define SOS_TOK VOCAB
#define T_DIM   256
#define B_DIM   32
#define D_DIM   512

// LDS B-panel column stride in halves: 512 + 8 pad -> 1040 B -> lane-to-lane
// advance of 4 banks; 16 lanes x b128 covers all 64 banks conflict-free.
#define BS_STRIDE 520

typedef __attribute__((ext_vector_type(16))) _Float16 v16h;
typedef __attribute__((ext_vector_type(8)))  _Float16 v8h;
typedef __attribute__((ext_vector_type(8)))  float    v8f;
typedef __attribute__((ext_vector_type(4)))  float    v4f;
typedef __attribute__((ext_vector_type(4)))  int      v4i;

// Address-space-qualified int4 pointers for the async-copy builtin
// (param 0 is AS1 v4i* per the compiler diagnostic; LDS side is AS3).
typedef __attribute__((address_space(1))) v4i as1_v4i;
typedef __attribute__((address_space(3))) v4i as3_v4i;

// ---- CDNA5 async global->LDS copy (ASYNCcnt path), with safe fallback ----
#if defined(__has_builtin)
#if __has_builtin(__builtin_amdgcn_global_load_async_to_lds_b128)
#define USE_ASYNC_COPY 1
#endif
#endif

#if defined(__has_builtin)
#if __has_builtin(__builtin_amdgcn_s_wait_asynccnt)
#define WAIT_ASYNC() __builtin_amdgcn_s_wait_asynccnt(0)
#endif
#endif
#ifndef WAIT_ASYNC
#define WAIT_ASYNC() asm volatile("s_wait_asynccnt 0x0" ::: "memory")
#endif

// ---------------------------------------------------------------------------
// Phase 1: encoder sum + decoder cumulative sum -> states, stored as f16.
// One thread per (b, d); b is uniform within a 256-thread block, so the
// t<len branches are wave-uniform.
// ---------------------------------------------------------------------------
__global__ void __launch_bounds__(256) states_kernel(
    const int*   __restrict__ inp,      // [T,B]
    const int*   __restrict__ in_len,   // [B]
    const int*   __restrict__ outp,     // [T,B]
    const int*   __restrict__ out_len,  // [B]
    const float* __restrict__ E,        // [VOCAB+1, D]
    _Float16*    __restrict__ Sh)       // [T*B, D] f16
{
    const int idx = blockIdx.x * blockDim.x + threadIdx.x; // b*D + d
    const int b = idx / D_DIM;
    const int d = idx - b * D_DIM;

    const int Lin = in_len[b];
    float acc = 0.0f;
    float g = 1.0f;
    for (int t = 0; t < Lin; ++t) {
        const int tok = inp[t * B_DIM + b];
        acc += g * E[(long long)tok * D_DIM + d];
        g *= GAMMA_F;
    }

    const int Lo = out_len[b];
    float run = acc;
    float g2 = 1.0f / GAMMA_F; // gamma^(t-1) at t=0
    for (int t = 0; t < T_DIM; ++t) {
        if (t < Lo) {
            const int tok = (t == 0) ? SOS_TOK : outp[(t - 1) * B_DIM + b];
            run -= g2 * E[(long long)tok * D_DIM + d];
        }
        g2 *= GAMMA_F;
        Sh[((long long)t * B_DIM + b) * D_DIM + d] = (_Float16)run;
    }
}

// ---------------------------------------------------------------------------
// Phase 2a: convert W (fp32 [VOCAB, D]) to f16 once. 8 elements per thread.
// ---------------------------------------------------------------------------
__global__ void __launch_bounds__(256) convw_kernel(
    const float* __restrict__ W,
    _Float16*    __restrict__ Wh)
{
    const long long i =
        ((long long)blockIdx.x * blockDim.x + threadIdx.x) * 8;
    v4f a = *(const v4f*)(W + i);
    v4f c = *(const v4f*)(W + i + 4);
    v8h o;
    o[0] = (_Float16)a[0]; o[1] = (_Float16)a[1];
    o[2] = (_Float16)a[2]; o[3] = (_Float16)a[3];
    o[4] = (_Float16)c[0]; o[5] = (_Float16)c[1];
    o[6] = (_Float16)c[2]; o[7] = (_Float16)c[3];
    *(v8h*)(Wh + i) = o;
}

// ---------------------------------------------------------------------------
// Phase 2b: GEMM  out[m, v] = states[m, :] . W[v, :] + bias[v]
// M=8192, N=32000, K=512.
//   * 512 threads = 16 waves per block.  Block tile: 256(M) x 64(N).
//   * The 64-column B panel (64x512 f16 = 64 KB) is staged into LDS once
//     (async global->LDS copies when available) and reused by all 16 waves;
//     WMMA B operands then come from conflict-free ds_load_b128.
//   * Per wave: 4 x v8f accumulators, 16 K-steps of wmma_f32_16x16x32_f16,
//     register double-buffered (A from global, B from LDS).
//   * Bias pre-folded into the accumulators (C layout: per-lane splat).
// Grid divides exactly -> no guards -> EXEC all ones at every WMMA.
// ---------------------------------------------------------------------------
__global__ void __launch_bounds__(512) gemm_kernel(
    const _Float16* __restrict__ Sh,   // [8192, 512] f16 (A, row-major)
    const _Float16* __restrict__ Wh,   // [32000, 512] f16 (B cols, K-contig)
    const float*    __restrict__ bias, // [32000]
    float*          __restrict__ out)  // [8192, 32000] f32
{
    __shared__ _Float16 Bs[64 * BS_STRIDE];   // 66,560 B

    const int tid  = threadIdx.x;
    const int lane = tid & 31;
    const int wave = tid >> 5;
    const int l16  = lane & 15;
    const int half = lane >> 4;

    const int ngrp  = blockIdx.x % 500;        // 0..499
    const int mgrp  = blockIdx.x / 500;        // 0..31
    const int mtile = mgrp * 16 + wave;        // 0..511
    const int n0    = ngrp * 64;
    const int mrow  = mtile * 16 + l16;

    // ---- stage B panel: 64 cols x 512 K halves -> LDS (column-major) ----
    // 4096 chunks of 16 B; 8 chunks per thread; global side coalesced.
    #pragma unroll
    for (int it = 0; it < 8; ++it) {
        const int chunk = it * 512 + tid;      // 0..4095
        const int col   = chunk >> 6;          // 0..63
        const int kc    = (chunk & 63) * 8;    // 0..504
        const _Float16* gsrc = Wh + (long long)(n0 + col) * D_DIM + kc;
        _Float16*       ldst = Bs + col * BS_STRIDE + kc;
#ifdef USE_ASYNC_COPY
        __builtin_amdgcn_global_load_async_to_lds_b128(
            (as1_v4i*)gsrc, (as3_v4i*)ldst, 0, 0);
#else
        *(v8h*)ldst = *(const v8h*)gsrc;
#endif
    }
#ifdef USE_ASYNC_COPY
    WAIT_ASYNC();
#endif
    __syncthreads();

    const _Float16* __restrict__ Arow = Sh + (long long)mrow * D_DIM;

    // Fold bias into accumulators: acc[j][r] covers (row = half*8+r,
    // col = n0+j*16+l16); bias depends only on col -> per-lane splat.
    v8f acc[4];
    #pragma unroll
    for (int j = 0; j < 4; ++j) {
        const float bb = bias[n0 + j * 16 + l16];
        #pragma unroll
        for (int r = 0; r < 8; ++r) acc[j][r] = bb;
    }

    // Fragment loaders.  A (global): lanes 0-15 K=k0+0..15, lanes
    // 16-31 K=k0+16..31.  B (LDS): col j*16+l16, 16 contiguous
    // halves at kb = k0 + half*16.
#define LOAD_A(dst, k0)                                                  \
    {   union { v16h v; v8h h[2]; } _u;                                  \
        _u.h[0] = *(const v8h*)(Arow + (k0) + half * 16);                \
        _u.h[1] = *(const v8h*)(Arow + (k0) + half * 16 + 8);            \
        dst = _u.v; }
#define LOAD_B(dst, j, kb)                                               \
    {   union { v16h v; v8h h[2]; } _u;                                  \
        const _Float16* _p = Bs + ((j) * 16 + l16) * BS_STRIDE + (kb);   \
        _u.h[0] = *(const v8h*)(_p);                                     \
        _u.h[1] = *(const v8h*)(_p + 8);                                 \
        dst = _u.v; }

    v16h a_cur, b0_cur, b1_cur, b2_cur, b3_cur;
    LOAD_A(a_cur, 0);
    LOAD_B(b0_cur, 0, half * 16);
    LOAD_B(b1_cur, 1, half * 16);
    LOAD_B(b2_cur, 2, half * 16);
    LOAD_B(b3_cur, 3, half * 16);

    #pragma unroll
    for (int k0 = 0; k0 < D_DIM; k0 += 32) {
        v16h a_nxt = a_cur, b0_nxt = b0_cur, b1_nxt = b1_cur,
             b2_nxt = b2_cur, b3_nxt = b3_cur;
        if (k0 + 32 < D_DIM) {
            const int kb = k0 + 32 + half * 16;
            LOAD_A(a_nxt, k0 + 32);
            LOAD_B(b0_nxt, 0, kb);
            LOAD_B(b1_nxt, 1, kb);
            LOAD_B(b2_nxt, 2, kb);
            LOAD_B(b3_nxt, 3, kb);
        }

        acc[0] = __builtin_amdgcn_wmma_f32_16x16x32_f16(
            false, a_cur, false, b0_cur, (short)0, acc[0], false, false);
        acc[1] = __builtin_amdgcn_wmma_f32_16x16x32_f16(
            false, a_cur, false, b1_cur, (short)0, acc[1], false, false);
        acc[2] = __builtin_amdgcn_wmma_f32_16x16x32_f16(
            false, a_cur, false, b2_cur, (short)0, acc[2], false, false);
        acc[3] = __builtin_amdgcn_wmma_f32_16x16x32_f16(
            false, a_cur, false, b3_cur, (short)0, acc[3], false, false);

        a_cur = a_nxt;
        b0_cur = b0_nxt; b1_cur = b1_nxt; b2_cur = b2_nxt; b3_cur = b3_nxt;
    }
#undef LOAD_A
#undef LOAD_B

    // C/D layout: lane -> column l16; VGPR r -> row half*8 + r.
    #pragma unroll
    for (int j = 0; j < 4; ++j) {
        const int col = n0 + j * 16 + l16;
        #pragma unroll
        for (int r = 0; r < 8; ++r) {
            const int row = mtile * 16 + half * 8 + r;
            out[(long long)row * VOCAB + col] = acc[j][r];
        }
    }
}

// ---------------------------------------------------------------------------
extern "C" void kernel_launch(void* const* d_in, const int* in_sizes, int n_in,
                              void* d_out, int out_size, void* d_ws,
                              size_t ws_size, hipStream_t stream) {
    const int*   inp     = (const int*)d_in[0];
    const int*   in_len  = (const int*)d_in[1];
    const int*   outp    = (const int*)d_in[2];
    const int*   out_len = (const int*)d_in[3];
    const float* E       = (const float*)d_in[4];
    const float* W       = (const float*)d_in[5];
    const float* bias    = (const float*)d_in[6];
    float*       out     = (float*)d_out;

    _Float16* Wh = (_Float16*)d_ws;                                  // 32.77 MB
    _Float16* Sh = (_Float16*)((char*)d_ws +
                               (size_t)VOCAB * D_DIM * sizeof(_Float16));

    // W fp32 -> f16 : 32000*512 / (256*8) = 8000 blocks (exact)
    convw_kernel<<<dim3(8000), dim3(256), 0, stream>>>(W, Wh);

    // states: B*D = 16384 threads = 64 blocks (exact)
    states_kernel<<<dim3(64), dim3(256), 0, stream>>>(
        inp, in_len, outp, out_len, E, Sh);

    // GEMM: (512/16) M-groups * 500 N-groups = 16000 blocks (exact)
    gemm_kernel<<<dim3(16000), dim3(512), 0, stream>>>(Sh, Wh, bias, out);
}
